// WeightGenerator_32478542692806
// MI455X (gfx1250) — matrix operand order repaired and checked by reading, not compiled, not run on hardware
//
#include <hip/hip_runtime.h>
#include <cmath>

typedef _Float16 v16h __attribute__((ext_vector_type(16)));
typedef _Float16 v8h  __attribute__((ext_vector_type(8)));
typedef float    v8f  __attribute__((ext_vector_type(8)));
typedef float    v4f  __attribute__((ext_vector_type(4)));

#define NN 1024
#define DD 32
#define HH 64
#define LN_EPS 1e-5f
#define AST 72   // LDS A-matrix row stride (halves)
#define WST 72   // LDS W2T row stride (halves)
#define HS  72   // LDS f16 hs/hd tile row stride (halves)

// ---------- Phase 1: hs = X @ W1[:32], hd = X @ W1[32:] ----------
__global__ __launch_bounds__(64) void pair_phase1(
    const float* __restrict__ X, const float* __restrict__ W1,
    float* __restrict__ hs, float* __restrict__ hd)
{
    const int row = blockIdx.x;
    const int col = threadIdx.x;
    float ss = 0.f, sd = 0.f;
    #pragma unroll 8
    for (int k = 0; k < DD; ++k) {
        float x = X[row * DD + k];
        ss = fmaf(x, W1[k * HH + col], ss);
        sd = fmaf(x, W1[(k + DD) * HH + col], sd);
    }
    hs[row * HH + col] = ss;
    hd[row * HH + col] = sd;
}

// ---------- Phase 1b: per-row scalars for LN1 stats decomposition ----------
// stats layout (floats): [0]=S_hs, [1024]=Q_hs, [2048]=P_hs (dot with b1),
//                        [3072]=S_hd, [4096]=Q_hd, [5120]=P_hd, [6144]=S_b1, [6145]=Q_b1
__global__ __launch_bounds__(256) void row_stats(
    const float* __restrict__ hs, const float* __restrict__ hd,
    const float* __restrict__ b1, float* __restrict__ stats)
{
    const int t = threadIdx.x;
    const int lane = t & 31;
    const int row = blockIdx.x * 8 + (t >> 5);
    const float b0 = b1[lane], bT = b1[lane + 32];

    float x0 = hs[row * HH + lane], x1 = hs[row * HH + lane + 32];
    float y0 = hd[row * HH + lane], y1 = hd[row * HH + lane + 32];
    float s  = x0 + x1,  q  = x0 * x0 + x1 * x1, p  = x0 * b0 + x1 * bT;
    float sd = y0 + y1,  qd = y0 * y0 + y1 * y1, pd = y0 * b0 + y1 * bT;
    #pragma unroll
    for (int m = 1; m < 32; m <<= 1) {
        s  += __shfl_xor(s, m, 32);  q  += __shfl_xor(q, m, 32);  p  += __shfl_xor(p, m, 32);
        sd += __shfl_xor(sd, m, 32); qd += __shfl_xor(qd, m, 32); pd += __shfl_xor(pd, m, 32);
    }
    if (lane == 0) {
        stats[row]        = s;  stats[1024 + row] = q;  stats[2048 + row] = p;
        stats[3072 + row] = sd; stats[4096 + row] = qd; stats[5120 + row] = pd;
    }
    if (blockIdx.x == 0 && t < 32) {
        float sb = b0 + bT, qb = b0 * b0 + bT * bT;
        #pragma unroll
        for (int m = 1; m < 32; m <<= 1) {
            sb += __shfl_xor(sb, m, 32); qb += __shfl_xor(qb, m, 32);
        }
        if (lane == 0) { stats[6144] = sb; stats[6145] = qb; }
    }
}

// ---------- Phase 2: WMMA dot-stats -> LN1 -> WMMA GEMM -> LN2 -> sigmoid ----------
__global__ __launch_bounds__(256) void pair_phase2(
    const float* __restrict__ hs, const float* __restrict__ hd,
    const float* __restrict__ stats,
    const float* __restrict__ b1, const float* __restrict__ g1, const float* __restrict__ be1,
    const float* __restrict__ W2,
    const float* __restrict__ g2, const float* __restrict__ be2, const float* __restrict__ b2,
    const float* __restrict__ W3, const float* __restrict__ b3,
    float* __restrict__ out)
{
    __shared__ __align__(16) float s_hs[16 * HH];
    __shared__ __align__(16) float s_hd[16 * HH];
    __shared__ __align__(16) _Float16 s_hsh[16 * HS];   // f16 copy, A-side of dot tile
    __shared__ __align__(16) _Float16 s_hdh[16 * HS];   // f16 copy, B-side of dot tile
    __shared__ __align__(16) float s_dot[16 * 16];      // dot(hs[i],hd[j]) tile
    __shared__ __align__(16) float s_b1[HH], s_g1[HH], s_be1[HH];
    __shared__ __align__(16) float s_b2[HH], s_g2[HH], s_be2[HH], s_w3[HH];
    __shared__ __align__(16) _Float16 s_A[256 * AST];
    __shared__ __align__(16) _Float16 s_W2T[HH * WST];

    const int t = threadIdx.x;
    const int ibase = blockIdx.y * 16;
    const int jbase = blockIdx.x * 16;
    const int lane = t & 31;
    const int lo = lane & 15;
    const int hi = lane >> 4;
    const int wid = t >> 5;

    // ---- stage ----
    for (int idx = t; idx < 16 * HH; idx += 256) {
        int r = idx >> 6, k = idx & 63;
        float vs = hs[(ibase + r) * HH + k];
        float vd = hd[(jbase + r) * HH + k];
        s_hs[idx] = vs;  s_hsh[r * HS + k] = (_Float16)vs;
        s_hd[idx] = vd;  s_hdh[r * HS + k] = (_Float16)vd;
    }
    if (t < HH) {
        s_b1[t] = b1[t]; s_g1[t] = g1[t]; s_be1[t] = be1[t];
        s_b2[t] = b2[t]; s_g2[t] = g2[t]; s_be2[t] = be2[t];
        s_w3[t] = W3[t];
    }
    for (int e = t; e < HH * HH; e += 256) {
        int k = e >> 6, n = e & 63;
        s_W2T[n * WST + k] = (_Float16)W2[e];
    }
    __syncthreads();

    // ---- dot(hs[i],hd[j]) tile via WMMA (all waves redundantly; EXEC stays full) ----
    {
        v8f dacc = {};
        #pragma unroll
        for (int kb = 0; kb < HH; kb += 32) {
            union { v16h v; v8h h[2]; } af, bf;
            const _Float16* ap = &s_hsh[lo * HS + kb + hi * 8];
            af.h[0] = *(const v8h*)(ap);
            af.h[1] = *(const v8h*)(ap + 16);
            const _Float16* bp = &s_hdh[lo * HS + kb + hi * 8];
            bf.h[0] = *(const v8h*)(bp);
            bf.h[1] = *(const v8h*)(bp + 16);
            dacc = __builtin_amdgcn_wmma_f32_16x16x32_f16(
                false, af.v, false, bf.v, (short)0, dacc, false, false);
        }
        #pragma unroll
        for (int r = 0; r < 8; ++r)
            s_dot[(hi * 8 + r) * 16 + lo] = dacc[r];   // identical values from all waves
    }
    __syncthreads();

    // ---- LN1: stats from decomposition, then materialize f16 A row ----
    {
        const int iloc = t >> 4, jloc = t & 15;
        const int gi = ibase + iloc, gj = jbase + jloc;
        const float SI = stats[gi],        QI = stats[1024 + gi], PI = stats[2048 + gi];
        const float SJ = stats[3072 + gj], QJ = stats[4096 + gj], PJ = stats[5120 + gj];
        const float Sb = stats[6144],      Qb = stats[6145];
        const float dot = s_dot[iloc * 16 + jloc];

        float sum = SI + SJ + Sb;
        float sq  = QI + QJ + Qb + 2.f * (dot + PI + PJ);
        float mu   = sum * (1.f / HH);
        float var  = sq * (1.f / HH) - mu * mu;
        float rstd = rsqrtf(var + LN_EPS);

        const float* hr = &s_hs[iloc * HH];
        const float* cr = &s_hd[jloc * HH];
        _Float16* arow = &s_A[t * AST];
        #pragma unroll
        for (int k = 0; k < HH; k += 4) {
            v4f a  = *(const v4f*)(hr + k);
            v4f c  = *(const v4f*)(cr + k);
            v4f bb = *(const v4f*)(s_b1 + k);
            v4f g  = *(const v4f*)(s_g1 + k);
            v4f be = *(const v4f*)(s_be1 + k);
            v4f v  = (a + c + bb - mu) * rstd * g + be;
            #pragma unroll
            for (int q = 0; q < 4; ++q) {
                float x = v[q] > 0.f ? v[q] : 0.f;
                arow[k + q] = (_Float16)x;
            }
        }
    }
    __syncthreads();

    // ---- hoist all 8 W2 B-fragments into registers (block-invariant) ----
    union { v16h v; v8h h[2]; } bw[8];
    #pragma unroll
    for (int kbi = 0; kbi < 2; ++kbi) {
        #pragma unroll
        for (int nt = 0; nt < 4; ++nt) {
            const _Float16* bp = &s_W2T[(nt * 16 + lo) * WST + kbi * 32 + hi * 8];
            bw[kbi * 4 + nt].h[0] = *(const v8h*)(bp);
            bw[kbi * 4 + nt].h[1] = *(const v8h*)(bp + 16);
        }
    }

    const float b3v = b3[0];

    #pragma unroll
    for (int halfIdx = 0; halfIdx < 2; ++halfIdx) {
        const int rt = wid + halfIdx * 8;     // row tile: i = ibase+rt
        const int rowbase = rt * 16;
        v8f acc[4] = {};

        #pragma unroll
        for (int kbi = 0; kbi < 2; ++kbi) {
            union { v16h v; v8h h[2]; } af;
            const _Float16* ap = &s_A[(rowbase + lo) * AST + kbi * 32 + hi * 8];
            af.h[0] = *(const v8h*)(ap);
            af.h[1] = *(const v8h*)(ap + 16);
            #pragma unroll
            for (int nt = 0; nt < 4; ++nt) {
                acc[nt] = __builtin_amdgcn_wmma_f32_16x16x32_f16(
                    false, af.v, false, bw[kbi * 4 + nt].v, (short)0, acc[nt], false, false);
            }
        }

        // LN2 + relu + W3 dot + sigmoid. acc[nt][r]: row hi*8+r, col nt*16+lo.
        float sOut[8];
        #pragma unroll
        for (int r = 0; r < 8; ++r) {
            float y[4];
            float s = 0.f, q = 0.f;
            #pragma unroll
            for (int nt = 0; nt < 4; ++nt) {
                int col = nt * 16 + lo;
                y[nt] = acc[nt][r] + s_b2[col];
                s += y[nt];
                q += y[nt] * y[nt];
            }
            #pragma unroll
            for (int m = 1; m < 16; m <<= 1) {
                s += __shfl_xor(s, m, 32);
                q += __shfl_xor(q, m, 32);
            }
            float mu   = s * (1.f / HH);
            float rstd = rsqrtf(q * (1.f / HH) - mu * mu + LN_EPS);
            float part = 0.f;
            #pragma unroll
            for (int nt = 0; nt < 4; ++nt) {
                int col = nt * 16 + lo;
                float x = (y[nt] - mu) * rstd * s_g2[col] + s_be2[col];
                x = x > 0.f ? x : 0.f;
                part = fmaf(x, s_w3[col], part);
            }
            #pragma unroll
            for (int m = 1; m < 16; m <<= 1) part += __shfl_xor(part, m, 32);
            sOut[r] = part;
        }

        if (lo == 0) {   // lane 0 -> j jbase..+7, lane 16 -> jbase+8..+15
            int i = ibase + rt;
            float* op = out + (size_t)i * NN + jbase + hi * 8;
            v4f o0, o1;
            #pragma unroll
            for (int r = 0; r < 4; ++r) o0[r] = 1.f / (1.f + expf(-(sOut[r] + b3v)));
            #pragma unroll
            for (int r = 0; r < 4; ++r) o1[r] = 1.f / (1.f + expf(-(sOut[4 + r] + b3v)));
            *(v4f*)(op)     = o0;
            *(v4f*)(op + 4) = o1;
        }
    }
}

extern "C" void kernel_launch(void* const* d_in, const int* in_sizes, int n_in,
                              void* d_out, int out_size, void* d_ws, size_t ws_size,
                              hipStream_t stream) {
    (void)in_sizes; (void)n_in; (void)out_size; (void)ws_size;
    const float* X   = (const float*)d_in[0];
    const float* W1  = (const float*)d_in[1];
    const float* b1  = (const float*)d_in[2];
    const float* g1  = (const float*)d_in[3];
    const float* be1 = (const float*)d_in[4];
    const float* W2  = (const float*)d_in[5];
    const float* b2  = (const float*)d_in[6];
    const float* g2  = (const float*)d_in[7];
    const float* be2 = (const float*)d_in[8];
    const float* W3  = (const float*)d_in[9];
    const float* b3  = (const float*)d_in[10];
    float* out = (float*)d_out;

    float* hs    = (float*)d_ws;                    // 1024*64 f32
    float* hd    = hs + (size_t)NN * HH;            // 1024*64 f32
    float* stats = hd + (size_t)NN * HH;            // 6*1024 + 2 f32

    pair_phase1<<<NN, HH, 0, stream>>>(X, W1, hs, hd);
    row_stats<<<NN / 8, 256, 0, stream>>>(hs, hd, b1, stats);
    dim3 grid(NN / 16, NN / 16);
    pair_phase2<<<grid, 256, 0, stream>>>(hs, hd, stats, b1, g1, be1, W2,
                                          g2, be2, b2, W3, b3, out);
}
